// ExogenousBasisLSTM_42528766165514
// MI455X (gfx1250) — matrix operand order repaired and checked by reading, not compiled, not run on hardware
//
#include <hip/hip_runtime.h>

// ---------------------------------------------------------------------------
// ExogenousBasisLSTM on MI455X (gfx1250, wave32, WMMA bf16 16x16x32)
// B=128, INF=64, OUTF=16, SEQ=512, FUT=96, H=512, L=2, PROJ=48
// v3: 1024-thread blocks (32 waves, 8/SIMD), 1 hidden ntile per wave,
// LICM defeated via laundered zero *offset* (keeps global addrspace ->
// global_load_b128 instead of flat_load_b128), split-K projection + LDS reduce.
// ---------------------------------------------------------------------------

typedef __attribute__((ext_vector_type(16))) __bf16 v16bf;
typedef __attribute__((ext_vector_type(8)))  float  v8f;

#define WMMA_BF16(a, b, c) \
  __builtin_amdgcn_wmma_f32_16x16x32_bf16(false, (a), false, (b), (short)0, (c), false, false)

__device__ __forceinline__ float sigf(float x) { return 1.0f / (1.0f + __expf(-x)); }

__device__ __forceinline__ v16bf load_bfrag(const __bf16* p) {
  return *(const v16bf*)p;  // 32B per lane, contiguous
}

// A-fragment for 16xK row-major bf16 matrix, row stride `stride`, K offset kOff.
// Lane l: m=l&15; g=(l>>4)*8; a[j]=X[m][kOff+g+j], a[8+j]=X[m][kOff+16+g+j]
__device__ __forceinline__ v16bf afrag(const __bf16* base, int stride, int lane, int kOff) {
  int m = lane & 15;
  int g = (lane >> 4) * 8;
  const __bf16* r = base + m * stride + kOff;
  v16bf a;
#pragma unroll
  for (int j = 0; j < 8; ++j) {
    a[j]     = r[g + j];
    a[8 + j] = r[16 + g + j];
  }
  return a;
}

// Defeat LICM without destroying address-space provenance: an opaque zero
// element offset, re-laundered every loop iteration.
__device__ __forceinline__ unsigned int zoff() {
  unsigned int z = 0;
  asm volatile("" : "+s"(z));
  return z;
}

// ---------------------------------------------------------------------------
// Packing kernels
// ---------------------------------------------------------------------------

// Pack W (Nreal x Kreal row-major, used as B = W^T) into B-fragment layout
// [Ktiles][Ntiles][32][16] bf16, zero padded.
__global__ void k_pack_w(const float* __restrict__ W, __bf16* __restrict__ out,
                         int Kreal, int Ktiles, int Nreal, int Ntiles) {
  int idx = blockIdx.x * 256 + threadIdx.x;
  int total = Ktiles * Ntiles * 32 * 16;
  if (idx >= total) return;
  int e    = idx & 15;
  int lane = (idx >> 4) & 31;
  int ktnt = idx >> 9;
  int nt   = ktnt % Ntiles;
  int kt   = ktnt / Ntiles;
  int k = kt * 32 + (lane >> 4) * 8 + (e & 7) + (e >> 3) * 16;
  int n = nt * 16 + (lane & 15);
  float v = (k < Kreal && n < Nreal) ? W[(size_t)n * Kreal + k] : 0.0f;
  out[idx] = (__bf16)v;
}

// insample_x_t (128, 64, 512) -> Xin [t][b][64] bf16
__global__ void k_pack_x(const float* __restrict__ x, __bf16* __restrict__ out) {
  int idx = blockIdx.x * 256 + threadIdx.x;
  if (idx >= 512 * 128 * 64) return;
  int f  = idx & 63;
  int tb = idx >> 6;
  int b  = tb & 127;
  int t  = tb >> 7;
  out[idx] = (__bf16)x[((size_t)b * 64 + f) * 512 + t];
}

// outsample_x_t (128, 16, 96) -> [t][b][16] bf16
__global__ void k_pack_decx(const float* __restrict__ x, __bf16* __restrict__ out) {
  int idx = blockIdx.x * 256 + threadIdx.x;
  if (idx >= 96 * 128 * 16) return;
  int f  = idx & 15;
  int tb = idx >> 4;
  int b  = tb & 127;
  int t  = tb >> 7;
  out[idx] = (__bf16)x[((size_t)b * 16 + f) * 96 + t];
}

// first_prev = insample[:, :48, -1] -> [b][64] bf16 zero padded
__global__ void k_first_prev(const float* __restrict__ x, __bf16* __restrict__ out) {
  int idx = blockIdx.x * 256 + threadIdx.x;
  if (idx >= 128 * 64) return;
  int f = idx & 63;
  int b = idx >> 6;
  out[idx] = (f < 48) ? (__bf16)x[((size_t)b * 64 + f) * 512 + 511] : (__bf16)0.0f;
}

__global__ void k_bias(const float* __restrict__ a, const float* __restrict__ b,
                       float* __restrict__ out) {
  int idx = blockIdx.x * 256 + threadIdx.x;
  if (idx < 2048) out[idx] = a[idx] + b[idx];
}

// ---------------------------------------------------------------------------
// Cell pieces (32-wave partition: wave w owns hidden ntile jt == w)
// ---------------------------------------------------------------------------

// Gates + pointwise for one hidden ntile. 16 WMMAs per wave.
__device__ __forceinline__ void cell_gates32(const __bf16* sIn, const __bf16* sH,
                                             const __bf16* WihB, const __bf16* WhhB,
                                             const float (&bv)[4],
                                             float (&c)[8], __bf16* sS,
                                             int wave, int lane) {
  v16bf ax0 = afrag(sIn, 64, lane, 0);
  v16bf ax1 = afrag(sIn, 64, lane, 32);
  v16bf ah0 = afrag(sH, 64, lane, 0);
  v16bf ah1 = afrag(sH, 64, lane, 32);
  int col  = lane & 15;
  int rowg = (lane >> 4) * 8;
  int jt   = wave;
  v8f acc[4];
#pragma unroll
  for (int q = 0; q < 4; ++q) {   // i, f, g, o quadrants
    int nt = q * 32 + jt;
    float b = bv[q];
    v8f a = {b, b, b, b, b, b, b, b};
    a = WMMA_BF16(ax0, load_bfrag(WihB + (((size_t)0 * 128 + nt) * 32 + lane) * 16), a);
    a = WMMA_BF16(ax1, load_bfrag(WihB + (((size_t)1 * 128 + nt) * 32 + lane) * 16), a);
    a = WMMA_BF16(ah0, load_bfrag(WhhB + (((size_t)0 * 128 + nt) * 32 + lane) * 16), a);
    a = WMMA_BF16(ah1, load_bfrag(WhhB + (((size_t)1 * 128 + nt) * 32 + lane) * 16), a);
    acc[q] = a;
  }
#pragma unroll
  for (int v = 0; v < 8; ++v) {
    float ig = sigf(acc[0][v]);
    float fg = sigf(acc[1][v]);
    float gg = tanhf(acc[2][v]);
    float og = sigf(acc[3][v]);
    float cn = fg * c[v] + ig * gg;
    c[v] = cn;
    sS[(v + rowg) * 512 + jt * 16 + col] = (__bf16)(og * tanhf(cn));
  }
}

// Projection split-K: waves 0..15, nt = w&3, k-quarter = w>>2 (4 ktiles each).
// Partials (f32) to pf[ (nt*4+kq) ][16][16].
__device__ __forceinline__ void cell_project_part(const __bf16* sS,
                                                  const __bf16* WhrB,
                                                  float* pf, int wave, int lane) {
  if (wave < 16) {
    int nt = wave & 3;
    int kq = wave >> 2;
    v8f acc = {};
#pragma unroll
    for (int k = 0; k < 4; ++k) {
      int kt = kq * 4 + k;
      v16bf a = afrag(sS, 512, lane, kt * 32);
      acc = WMMA_BF16(a, load_bfrag(WhrB + (((size_t)kt * 4 + nt) * 32 + lane) * 16), acc);
    }
    int col  = lane & 15;
    int rowg = (lane >> 4) * 8;
#pragma unroll
    for (int v = 0; v < 8; ++v)
      pf[((nt * 4 + kq) * 16 + (v + rowg)) * 16 + col] = acc[v];
  }
}

// Combine 4 partials per output element; write sHout [16][64] (+ optional
// global copies, same layout). 1024 threads, one element each.
__device__ __forceinline__ void cell_project_combine(const float* pf, __bf16* sHout,
                                                     __bf16* gOut, __bf16* gOut2,
                                                     int tid) {
  int row = tid >> 6;
  int cc  = tid & 63;
  int nt  = cc >> 4;
  int col = cc & 15;
  const float* q = pf + (nt * 4) * 256 + row * 16 + col;
  float s = q[0] + q[256] + q[512] + q[768];
  __bf16 hv = (__bf16)s;
  sHout[tid] = hv;
  if (gOut)  gOut[tid]  = hv;
  if (gOut2) gOut2[tid] = hv;
}

// ---------------------------------------------------------------------------
// Encoder layer: one block (1024 thr, 32 waves) per batch tile of 16 rows.
// Xin/Hseq: [T][8][16][64] bf16. hT: [8][16][64] bf16. cT: [8][16][512] f32.
// ---------------------------------------------------------------------------
__global__ void __launch_bounds__(1024)
k_lstm_layer(const __bf16* __restrict__ Xin, const __bf16* __restrict__ WihB,
             const __bf16* __restrict__ WhhB, const __bf16* __restrict__ WhrB,
             const float* __restrict__ bias, __bf16* __restrict__ Hseq,
             __bf16* __restrict__ hT, float* __restrict__ cT, int T) {
  __shared__ __bf16 sH[16 * 64];
  __shared__ __bf16 sS[16 * 512];
  __shared__ float  pf[16 * 256];
  int bt = blockIdx.x;
  int tid = threadIdx.x;
  int wave = tid >> 5, lane = tid & 31;
  int col = lane & 15, rowg = (lane >> 4) * 8;
  sH[tid] = (__bf16)0.0f;
  float c[8];
#pragma unroll
  for (int v = 0; v < 8; ++v) c[v] = 0.0f;
  float bv[4];
#pragma unroll
  for (int q = 0; q < 4; ++q) bv[q] = bias[(q * 32 + wave) * 16 + col];

  for (int t = 0; t < T; ++t) {
    __syncthreads();  // sH from previous combine visible
    unsigned int z = zoff();
    const __bf16* wih = WihB + z;
    const __bf16* whh = WhhB + z;
    const __bf16* whr = WhrB + z;
    const __bf16* xin = Xin + (size_t)(t * 8 + bt) * 1024;
    cell_gates32(xin, sH, wih, whh, bv, c, sS, wave, lane);
    __syncthreads();  // sS complete; sH reads done
    cell_project_part(sS, whr, pf, wave, lane);
    __syncthreads();  // pf complete
    cell_project_combine(pf, sH, Hseq + (size_t)(t * 8 + bt) * 1024, nullptr, tid);
  }
  __syncthreads();
#pragma unroll
  for (int v = 0; v < 8; ++v)
    cT[((size_t)bt * 16 + v + rowg) * 512 + wave * 16 + col] = c[v];
  hT[(size_t)bt * 1024 + tid] = sH[tid];
}

// ---------------------------------------------------------------------------
// Decoder: both layers fused, 96 steps, prev-feedback through LDS.
// ---------------------------------------------------------------------------
__global__ void __launch_bounds__(1024)
k_lstm_dec(const __bf16* __restrict__ DecX,       // [96][8][16][16]
           const __bf16* __restrict__ FirstPrev,  // [8][16][64]
           const __bf16* __restrict__ hT,         // [2][8][16][64]
           const float*  __restrict__ cT,         // [2][8][16][512]
           const __bf16* __restrict__ WihB0, const __bf16* __restrict__ WhhB0,
           const __bf16* __restrict__ WhrB0, const float* __restrict__ bias0,
           const __bf16* __restrict__ WihB1, const __bf16* __restrict__ WhhB1,
           const __bf16* __restrict__ WhrB1, const float* __restrict__ bias1,
           __bf16* __restrict__ Pred,             // [96][8][16][64]
           int T) {
  __shared__ __bf16 sIn[16 * 64];
  __shared__ __bf16 sH0[16 * 64];
  __shared__ __bf16 sH1[16 * 64];
  __shared__ __bf16 sPrev[16 * 64];
  __shared__ __bf16 sS[16 * 512];
  __shared__ float  pf[16 * 256];
  int bt = blockIdx.x, tid = threadIdx.x;
  int wave = tid >> 5, lane = tid & 31;
  int col = lane & 15, rowg = (lane >> 4) * 8;

  sPrev[tid] = FirstPrev[(size_t)bt * 1024 + tid];
  sH0[tid]   = hT[(size_t)bt * 1024 + tid];
  sH1[tid]   = hT[(size_t)8 * 1024 + bt * 1024 + tid];

  float c0[8], c1[8];
#pragma unroll
  for (int v = 0; v < 8; ++v) {
    size_t base = ((size_t)bt * 16 + v + rowg) * 512 + wave * 16 + col;
    c0[v] = cT[base];
    c1[v] = cT[(size_t)8 * 16 * 512 + base];
  }
  float bv0[4], bv1[4];
#pragma unroll
  for (int q = 0; q < 4; ++q) {
    bv0[q] = bias0[(q * 32 + wave) * 16 + col];
    bv1[q] = bias1[(q * 32 + wave) * 16 + col];
  }

  for (int t = 0; t < T; ++t) {
    __syncthreads();  // sH1/sPrev from previous step visible
    {
      int m = tid >> 6, f = tid & 63;
      sIn[tid] = (f < 48) ? sPrev[m * 64 + f]
                          : DecX[((size_t)(t * 8 + bt) * 16 + m) * 16 + (f - 48)];
    }
    __syncthreads();
    unsigned int z = zoff();
    cell_gates32(sIn, sH0, WihB0 + z, WhhB0 + z, bv0, c0, sS, wave, lane);
    __syncthreads();
    cell_project_part(sS, WhrB0 + z, pf, wave, lane);
    __syncthreads();
    cell_project_combine(pf, sH0, nullptr, nullptr, tid);
    __syncthreads();
    unsigned int z2 = zoff();
    cell_gates32(sH0, sH1, WihB1 + z2, WhhB1 + z2, bv1, c1, sS, wave, lane);
    __syncthreads();
    cell_project_part(sS, WhrB1 + z2, pf, wave, lane);
    __syncthreads();
    cell_project_combine(pf, sH1, sPrev,
                         Pred + (size_t)(t * 8 + bt) * 1024, tid);
  }
}

// ---------------------------------------------------------------------------
// Final einsums: backcast[b,t] = theta[b,48:]·Hseq1[b,t,:48]
//                forecast[b,t] = theta[b,:48]·Pred[b,t,:48]
// ---------------------------------------------------------------------------
__global__ void k_einsum(const float* __restrict__ theta,
                         const __bf16* __restrict__ Hseq1,
                         const __bf16* __restrict__ Pred,
                         float* __restrict__ out) {
  int idx = blockIdx.x * 256 + threadIdx.x;
  const int totalB = 128 * 512;
  if (idx < totalB) {
    int b = idx >> 9, t = idx & 511;
    const __bf16* row = Hseq1 + ((size_t)t * 128 + b) * 64;
    const float* th = theta + b * 96 + 48;
    float acc = 0.0f;
#pragma unroll
    for (int p = 0; p < 48; ++p) acc += th[p] * (float)row[p];
    out[idx] = acc;
  } else if (idx < totalB + 128 * 96) {
    int j = idx - totalB;
    int b = j / 96, t = j % 96;
    const __bf16* row = Pred + ((size_t)t * 128 + b) * 64;
    const float* th = theta + b * 96;
    float acc = 0.0f;
#pragma unroll
    for (int p = 0; p < 48; ++p) acc += th[p] * (float)row[p];
    out[totalB + j] = acc;
  }
}

// ---------------------------------------------------------------------------
// Host launcher
// ---------------------------------------------------------------------------
static inline int nblk(long n) { return (int)((n + 255) / 256); }

extern "C" void kernel_launch(void* const* d_in, const int* in_sizes, int n_in,
                              void* d_out, int out_size, void* d_ws, size_t ws_size,
                              hipStream_t stream) {
  (void)in_sizes; (void)n_in; (void)out_size; (void)ws_size;
  const float* theta = (const float*)d_in[0];
  const float* xenc  = (const float*)d_in[1];
  const float* xdec  = (const float*)d_in[2];
  // layers: 0=enc0, 1=enc1, 2=dec0, 3=dec1 at input indices 3,8,13,18
  const float *Wih[4], *Whh[4], *bih[4], *bhh[4], *Whr[4];
  for (int l = 0; l < 4; ++l) {
    int base = 3 + l * 5;
    Wih[l] = (const float*)d_in[base + 0];
    Whh[l] = (const float*)d_in[base + 1];
    bih[l] = (const float*)d_in[base + 2];
    bhh[l] = (const float*)d_in[base + 3];
    Whr[l] = (const float*)d_in[base + 4];
  }
  const int KihReal[4] = {64, 48, 64, 48};

  // ---- carve workspace ----
  char* p = (char*)d_ws;
  auto take = [&](size_t bytes) -> char* {
    char* r = p;
    p += (bytes + 255) & ~(size_t)255;
    return r;
  };
  __bf16* Xin0   = (__bf16*)take((size_t)512 * 128 * 64 * 2);
  __bf16* Hseq0  = (__bf16*)take((size_t)512 * 128 * 64 * 2);
  __bf16* Hseq1  = (__bf16*)take((size_t)512 * 128 * 64 * 2);
  __bf16* DecX   = (__bf16*)take((size_t)96 * 128 * 16 * 2);
  __bf16* Fprev  = (__bf16*)take((size_t)128 * 64 * 2);
  __bf16* hTb    = (__bf16*)take((size_t)2 * 128 * 64 * 2);
  float*  cTb    = (float*)take((size_t)2 * 128 * 512 * 4);
  __bf16* Pred   = (__bf16*)take((size_t)96 * 128 * 64 * 2);
  __bf16* WihB[4]; __bf16* WhhB[4]; __bf16* WhrB[4]; float* biasS[4];
  for (int l = 0; l < 4; ++l) {
    WihB[l]  = (__bf16*)take((size_t)2 * 128 * 32 * 16 * 2);
    WhhB[l]  = (__bf16*)take((size_t)2 * 128 * 32 * 16 * 2);
    WhrB[l]  = (__bf16*)take((size_t)16 * 4 * 32 * 16 * 2);  // 4th ntile = zeros
    biasS[l] = (float*)take((size_t)2048 * 4);
  }

  // ---- pack inputs & weights ----
  k_pack_x<<<nblk(512L * 128 * 64), 256, 0, stream>>>(xenc, Xin0);
  k_pack_decx<<<nblk(96L * 128 * 16), 256, 0, stream>>>(xdec, DecX);
  k_first_prev<<<nblk(128L * 64), 256, 0, stream>>>(xenc, Fprev);
  for (int l = 0; l < 4; ++l) {
    k_pack_w<<<nblk(2L * 128 * 512), 256, 0, stream>>>(Wih[l], WihB[l], KihReal[l], 2, 2048, 128);
    k_pack_w<<<nblk(2L * 128 * 512), 256, 0, stream>>>(Whh[l], WhhB[l], 48, 2, 2048, 128);
    k_pack_w<<<nblk(16L * 4 * 512), 256, 0, stream>>>(Whr[l], WhrB[l], 512, 16, 48, 4);
    k_bias<<<8, 256, 0, stream>>>(bih[l], bhh[l], biasS[l]);
  }

  // ---- encoder (2 layers, sequential) ----
  k_lstm_layer<<<8, 1024, 0, stream>>>(Xin0, WihB[0], WhhB[0], WhrB[0], biasS[0],
                                       Hseq0, hTb, cTb, 512);
  k_lstm_layer<<<8, 1024, 0, stream>>>(Hseq0, WihB[1], WhhB[1], WhrB[1], biasS[1],
                                       Hseq1, hTb + (size_t)128 * 64,
                                       cTb + (size_t)128 * 512, 512);

  // ---- decoder (2 fused layers, 96 steps) ----
  k_lstm_dec<<<8, 1024, 0, stream>>>(DecX, Fprev, hTb, cTb,
                                     WihB[2], WhhB[2], WhrB[2], biasS[2],
                                     WihB[3], WhhB[3], WhrB[3], biasS[3],
                                     Pred, 96);

  // ---- final einsums -> d_out ----
  k_einsum<<<nblk(128L * 512 + 128L * 96), 256, 0, stream>>>(theta, Hseq1, Pred,
                                                             (float*)d_out);
}
